// SAAttnMem_3891240370915
// MI455X (gfx1250) — matrix operand order repaired and checked by reading, not compiled, not run on hardware
//
#include <hip/hip_runtime.h>
#include <hip/hip_bf16.h>
#include <math.h>

typedef _Float16 h16;
typedef __attribute__((ext_vector_type(16))) _Float16 v16h;
typedef __attribute__((ext_vector_type(8)))  float    v8f;

#define NB 8
#define SS 4096   // H*W = 64*64

union AFrag {
  v16h v;
  h16  h[16];
  unsigned u[8];
};

// ---- WMMA wrapper: D = A(16x32 f16) x B(32x16 f16) + C(16x16 f32) ----
__device__ inline v8f wmma16(const AFrag& a, const AFrag& b, v8f c) {
  return __builtin_amdgcn_wmma_f32_16x16x32_f16(
      false, a.v, false, b.v, (short)0, c, false, false);
}

// DPP lane-permute within 16-lane rows (VALU, no LDS traffic)
template <int CTRL>
__device__ inline float dppmov(float x) {
  int v = __float_as_int(x);
  v = __builtin_amdgcn_update_dpp(v, v, CTRL, 0xF, 0xF, false);
  return __int_as_float(v);
}
// butterfly max across the 16 lanes of a row: xor1, xor2, xor7, xor15
__device__ inline float rowmax16(float x) {
  x = fmaxf(x, dppmov<0xB1>(x));   // quad_perm [1,0,3,2]  == xor 1
  x = fmaxf(x, dppmov<0x4E>(x));   // quad_perm [2,3,0,1]  == xor 2
  x = fmaxf(x, dppmov<0x141>(x));  // row_half_mirror      == xor 7
  x = fmaxf(x, dppmov<0x140>(x));  // row_mirror           == xor 15
  return x;
}

// A-matrix 16x32 from row-major f16 [row][k], row stride ld (halfs, mult of 8).
__device__ inline void loadA_v(AFrag& a, const h16* base, int ld, int lane) {
  int row = lane & 15, hi = lane >> 4;
  const h16* p = base + (size_t)row * ld + 8 * hi;
  *(uint4*)&a.u[0] = *(const uint4*)p;
  *(uint4*)&a.u[4] = *(const uint4*)(p + 16);
}

// B-matrix 32x16 from K-contiguous source src[n][k], row stride ld (mult of 8).
__device__ inline void loadB_v(AFrag& b, const h16* base, int ld, int lane) {
  int col = lane & 15, hi = lane >> 4;
  const h16* p = base + (size_t)col * ld + 8 * hi;
  *(uint4*)&b.u[0] = *(const uint4*)p;
  *(uint4*)&b.u[4] = *(const uint4*)(p + 16);
}

// ---------------------------------------------------------------------------
// Kernel 0: weight prep -> f16, conv3x3 weights reordered to k=(ky,kx,c)
// ---------------------------------------------------------------------------
__global__ __launch_bounds__(256)
void wprep_kernel(const float* __restrict__ Wz, const float* __restrict__ Wo,
                  h16* __restrict__ WzH, h16* __restrict__ WoR)
{
  int idx = blockIdx.x * 256 + threadIdx.x;
  if (idx < 64 * 128) WzH[idx] = (h16)Wz[idx];
  if (idx < 192 * 1152) {
    int o = idx / 1152;
    int rem = idx - o * 1152;
    int r = rem >> 7;          // tap index (ky*3+kx)
    int c = rem & 127;         // input channel
    WoR[idx] = (h16)Wo[(size_t)o * 1152 + c * 9 + r];
  }
}

// ---------------------------------------------------------------------------
// Kernel 1: QKV/KV projections. q is pre-scaled by 0.125*log2(e) so the
// attention kernel can use v_exp_f32 (exp2) directly with no score scaling.
//   qT/khT/kmT (N,S,64)  [c contiguous]   vhB/vmB (N,64,S)  [t contiguous]
//   xT (N,S,128) f16: channels 64..127 = h
// ---------------------------------------------------------------------------
__global__ __launch_bounds__(256)
void proj_kernel(const float* __restrict__ hIn, const float* __restrict__ mIn,
                 const float* __restrict__ Wh, const float* __restrict__ bh,
                 const float* __restrict__ Wm, const float* __restrict__ bm,
                 h16* __restrict__ qT, h16* __restrict__ khT, h16* __restrict__ vhB,
                 h16* __restrict__ kmT, h16* __restrict__ vmB, h16* __restrict__ xT)
{
  const float QSCALE = 0.18033688011112042f;  // (1/8) * log2(e)
  __shared__ float wf[192 * 64];
  int tid = threadIdx.x;
  int n = blockIdx.x >> 4;
  int s = (blockIdx.x & 15) * 256 + tid;

  float in[64];
  union { h16 hh[64]; uint4 q4[8]; } seg;

  // ---- phase 1: h projections (192 outputs) ----
  for (int i = tid; i < 192 * 64; i += 256) wf[i] = Wh[i];
  __syncthreads();
  #pragma unroll
  for (int c = 0; c < 64; ++c) {
    in[c] = hIn[((size_t)n * 64 + c) * SS + s];
    seg.hh[c] = (h16)in[c];
  }
  {
    uint4* dst = (uint4*)(xT + ((size_t)n * SS + s) * 128 + 64);
    #pragma unroll
    for (int j = 0; j < 8; ++j) dst[j] = seg.q4[j];
  }
  #pragma unroll
  for (int blk = 0; blk < 3; ++blk) {      // q / k / v
    #pragma unroll 4
    for (int oo = 0; oo < 64; ++oo) {
      int o = blk * 64 + oo;
      float acc = bh[o];
      const float* wr = &wf[o * 64];
      #pragma unroll
      for (int c = 0; c < 64; ++c) acc += wr[c] * in[c];
      if (blk == 0) acc *= QSCALE;
      seg.hh[oo] = (h16)acc;
    }
    if (blk == 0) {
      uint4* dst = (uint4*)(qT + ((size_t)n * SS + s) * 64);
      #pragma unroll
      for (int j = 0; j < 8; ++j) dst[j] = seg.q4[j];
    } else if (blk == 1) {
      uint4* dst = (uint4*)(khT + ((size_t)n * SS + s) * 64);
      #pragma unroll
      for (int j = 0; j < 8; ++j) dst[j] = seg.q4[j];
    } else {
      #pragma unroll 8
      for (int c = 0; c < 64; ++c)
        vhB[(size_t)n * 64 * SS + (size_t)c * SS + s] = seg.hh[c];
    }
  }
  __syncthreads();

  // ---- phase 2: m projections (128 outputs) ----
  for (int i = tid; i < 128 * 64; i += 256) wf[i] = Wm[i];
  __syncthreads();
  #pragma unroll
  for (int c = 0; c < 64; ++c) in[c] = mIn[((size_t)n * 64 + c) * SS + s];
  #pragma unroll
  for (int blk = 0; blk < 2; ++blk) {      // k / v
    #pragma unroll 4
    for (int oo = 0; oo < 64; ++oo) {
      int o = blk * 64 + oo;
      float acc = bm[o];
      const float* wr = &wf[o * 64];
      #pragma unroll
      for (int c = 0; c < 64; ++c) acc += wr[c] * in[c];
      seg.hh[oo] = (h16)acc;
    }
    if (blk == 0) {
      uint4* dst = (uint4*)(kmT + ((size_t)n * SS + s) * 64);
      #pragma unroll
      for (int j = 0; j < 8; ++j) dst[j] = seg.q4[j];
    } else {
      #pragma unroll 8
      for (int c = 0; c < 64; ++c)
        vmB[(size_t)n * 64 * SS + (size_t)c * SS + s] = seg.hh[c];
    }
  }
}

// ---------------------------------------------------------------------------
// Kernel 2: flash attention. Block = 4 waves, wave = 16-query tile.
// 64-key chunks, double-buffered LDS, ONE barrier per chunk, and a split
// K-then-V prefetch pipeline so only 4 staging uint4s are ever live
// (avoids the VGPR spills seen with combined K+V staging).
// Per chunk: 8 score WMMAs -> DPP row-max + exp2 -> probs via LDS ->
// 2 rowsum WMMAs (P x ones) + 8 AV WMMAs. Writes ZcatT (N,S,128).
// ---------------------------------------------------------------------------
__global__ __launch_bounds__(128)
void attn_kernel(const h16* __restrict__ qT, const h16* __restrict__ khT,
                 const h16* __restrict__ vhB, const h16* __restrict__ kmT,
                 const h16* __restrict__ vmB, h16* __restrict__ ZT)
{
  __shared__ h16 kt[2][64 * 72];    // K chunk: 64 keys x 64 c (+pad)
  __shared__ h16 vc[2][64 * 72];    // V chunk: 64 c x 64 keys (+pad)
  __shared__ h16 pst[4 * 16 * 72];  // per-wave probs 16 x 64 (+pad)

  int tid = threadIdx.x;
  int wave = tid >> 5, lane = tid & 31;
  int col = lane & 15, hi = lane >> 4;

  int bid = blockIdx.x;
  int qg = bid & 63;
  int which = (bid >> 6) & 1;
  int n = bid >> 7;

  const h16* kG = (which ? kmT : khT) + (size_t)n * SS * 64;   // (S,64)
  const h16* vG = (which ? vmB : vhB) + (size_t)n * 64 * SS;   // (64,S)
  const h16* qG = qT + (size_t)n * SS * 64;

  int s0 = qg * 64 + wave * 16;

  AFrag aq0, aq1;  // q fragments (pre-scaled), contraction c = 0..63
  loadA_v(aq0, qG + (size_t)s0 * 64,      64, lane);
  loadA_v(aq1, qG + (size_t)s0 * 64 + 32, 64, lane);

  AFrag bones;     // all-ones B fragment for row sums
  #pragma unroll
  for (int i = 0; i < 8; ++i) bones.u[i] = 0x3C003C00u;

  float rowm[8];
  v8f z[4];
  v8f lsum = (v8f){};
  #pragma unroll
  for (int j = 0; j < 8; ++j) rowm[j] = -1e30f;
  #pragma unroll
  for (int ct = 0; ct < 4; ++ct) z[ct] = (v8f){};

  h16* pmy = &pst[wave * 16 * 72];

  // staging: 128 threads, each moves half a 64-half row of K and of V (64B)
  int rk = tid >> 1;              // row 0..63
  int pk = (tid & 1) * 32;        // 32-half part

  { // prologue: stage chunk 0 into buffer 0 (K then V, reusing registers)
    uint4 st[4];
    const uint4* g = (const uint4*)(kG + (size_t)rk * 64 + pk);
    st[0] = g[0]; st[1] = g[1]; st[2] = g[2]; st[3] = g[3];
    uint4* lk = (uint4*)&kt[0][rk * 72 + pk];
    lk[0] = st[0]; lk[1] = st[1]; lk[2] = st[2]; lk[3] = st[3];
    const uint4* v = (const uint4*)(vG + (size_t)rk * SS + pk);
    st[0] = v[0]; st[1] = v[1]; st[2] = v[2]; st[3] = v[3];
    uint4* lv = (uint4*)&vc[0][rk * 72 + pk];
    lv[0] = st[0]; lv[1] = st[1]; lv[2] = st[2]; lv[3] = st[3];
  }

  for (int it = 0; it < SS / 64; ++it) {
    __syncthreads();                    // buf[cur] ready; buf[cur^1] free
    int cur = it & 1, nb = cur ^ 1;
    bool more = (it + 1) < SS / 64;
    int t0n = (it + 1) * 64;

    uint4 st[4];
    if (more) {                         // issue next chunk's K loads
      const uint4* g = (const uint4*)(kG + (size_t)(t0n + rk) * 64 + pk);
      st[0] = g[0]; st[1] = g[1]; st[2] = g[2]; st[3] = g[3];
    }
    const h16* KT = kt[cur];
    const h16* VC = vc[cur];

    // scores: four 16(query)x16(key) tiles (log2 units via q pre-scale)
    v8f sc[4];
    #pragma unroll
    for (int nn = 0; nn < 4; ++nn) {
      v8f s = (v8f){};
      AFrag b;
      loadB_v(b, KT + (size_t)(16 * nn) * 72,      72, lane);
      s = wmma16(aq0, b, s);
      loadB_v(b, KT + (size_t)(16 * nn) * 72 + 32, 72, lane);
      s = wmma16(aq1, b, s);
      sc[nn] = s;
    }

    if (more) {                         // commit K, then issue V loads
      uint4* lk = (uint4*)&kt[nb][rk * 72 + pk];
      lk[0] = st[0]; lk[1] = st[1]; lk[2] = st[2]; lk[3] = st[3];
      const uint4* v = (const uint4*)(vG + (size_t)rk * SS + t0n + pk);
      st[0] = v[0]; st[1] = v[1]; st[2] = v[2]; st[3] = v[3];
    }

    // online softmax: DPP row-max over 64 keys, exp2, accumulator rescale
    #pragma unroll
    for (int j = 0; j < 8; ++j) {
      float x0 = sc[0][j], x1 = sc[1][j], x2 = sc[2][j], x3 = sc[3][j];
      float mx = rowmax16(fmaxf(fmaxf(x0, x1), fmaxf(x2, x3)));
      float mnew = fmaxf(rowm[j], mx);
      float corr = __builtin_amdgcn_exp2f(rowm[j] - mnew);
      rowm[j] = mnew;
      lsum[j] *= corr;
      #pragma unroll
      for (int ct = 0; ct < 4; ++ct) z[ct][j] *= corr;
      int prow = 8 * hi + j;
      pmy[prow * 72 +      col] = (h16)__builtin_amdgcn_exp2f(x0 - mnew);
      pmy[prow * 72 + 16 + col] = (h16)__builtin_amdgcn_exp2f(x1 - mnew);
      pmy[prow * 72 + 32 + col] = (h16)__builtin_amdgcn_exp2f(x2 - mnew);
      pmy[prow * 72 + 48 + col] = (h16)__builtin_amdgcn_exp2f(x3 - mnew);
    }

    // row sums via P x ones; AV per channel tile (in-wave LDS ordering)
    AFrag ap0, ap1;
    loadA_v(ap0, pmy,      72, lane);
    loadA_v(ap1, pmy + 32, 72, lane);
    lsum = wmma16(ap0, bones, lsum);
    lsum = wmma16(ap1, bones, lsum);
    #pragma unroll
    for (int ct = 0; ct < 4; ++ct) {
      AFrag bv;
      loadB_v(bv, VC + (size_t)(16 * ct) * 72,      72, lane);
      z[ct] = wmma16(ap0, bv, z[ct]);
      loadB_v(bv, VC + (size_t)(16 * ct) * 72 + 32, 72, lane);
      z[ct] = wmma16(ap1, bv, z[ct]);
    }

    if (more) {                         // commit V
      uint4* lv = (uint4*)&vc[nb][rk * 72 + pk];
      lv[0] = st[0]; lv[1] = st[1]; lv[2] = st[2]; lv[3] = st[3];
    }
  }

  // epilogue: normalize by rowsum, write ZcatT (S,128) channel-contiguous
  #pragma unroll
  for (int ct = 0; ct < 4; ++ct) {
    int ch = which * 64 + 16 * ct + col;
    #pragma unroll
    for (int j = 0; j < 8; ++j) {
      int s = s0 + 8 * hi + j;
      ZT[((size_t)n * SS + s) * 128 + ch] = (h16)(z[ct][j] / lsum[j]);
    }
  }
}

// ---------------------------------------------------------------------------
// Kernel 3: conv_z 1x1 (64 <- 128) via WMMA; writes Z into xT channels 0..63.
// ---------------------------------------------------------------------------
__global__ __launch_bounds__(128)
void convz_kernel(const h16* __restrict__ WzH, const float* __restrict__ bz,
                  const h16* __restrict__ ZT, h16* __restrict__ xT)
{
  int lane = threadIdx.x & 31;
  int wave = threadIdx.x >> 5;
  int t = blockIdx.x * 4 + wave;
  int st = t & 255, ot = (t >> 8) & 3, n = t >> 10;
  int s0 = st * 16, o0 = ot * 16;
  int col = lane & 15, hi = lane >> 4;

  const h16* Zg = ZT + (size_t)n * SS * 128 + (size_t)s0 * 128;
  v8f acc = (v8f){};
  #pragma unroll
  for (int kk = 0; kk < 4; ++kk) {
    AFrag a, b;
    loadA_v(a, WzH + (size_t)o0 * 128 + 32 * kk, 128, lane);
    loadB_v(b, Zg + 32 * kk, 128, lane);
    acc = wmma16(a, b, acc);
  }
  union { h16 hh[8]; uint4 q4; } pk;
  #pragma unroll
  for (int j = 0; j < 8; ++j) pk.hh[j] = (h16)(acc[j] + bz[o0 + 8 * hi + j]);
  *(uint4*)(xT + ((size_t)n * SS + s0 + col) * 128 + o0 + 8 * hi) = pk.q4;
}

// ---------------------------------------------------------------------------
// Kernel 4: 3x3 conv (192 <- 128) implicit GEMM, K=(ky,kx,c)=1152, 36 steps.
// ---------------------------------------------------------------------------
__global__ __launch_bounds__(128)
void conv3_kernel(const h16* __restrict__ WoR, const float* __restrict__ bo,
                  const h16* __restrict__ xT, float* __restrict__ yws)
{
  int lane = threadIdx.x & 31;
  int wave = threadIdx.x >> 5;
  int t = blockIdx.x * 4 + wave;
  int pt = t & 255, ot = (t >> 8) % 12, n = t / 3072;
  int s0 = pt * 16, o0 = ot * 16;
  int col = lane & 15, hi = lane >> 4;
  int y = s0 >> 6, x0 = s0 & 63;

  const h16* xg = xT + (size_t)n * SS * 128;
  v8f acc = (v8f){};
  for (int kk = 0; kk < 36; ++kk) {
    int kb = 32 * kk;
    int r = kb >> 7;
    int cb = kb & 127;
    int ky = r / 3 - 1, kx = r - (r / 3) * 3 - 1;
    int yy = y + ky;
    int xx = x0 + col + kx;

    AFrag a;
    loadA_v(a, WoR + (size_t)o0 * 1152 + kb, 1152, lane);

    uint4 b0{}, b1{};
    if ((unsigned)yy < 64u && (unsigned)xx < 64u) {
      const uint4* p = (const uint4*)(xg + ((size_t)(yy * 64 + xx)) * 128 + cb + 8 * hi);
      b0 = p[0]; b1 = p[2];
    }
    AFrag b;
    *(uint4*)&b.u[0] = b0;
    *(uint4*)&b.u[4] = b1;
    acc = wmma16(a, b, acc);
  }
  #pragma unroll
  for (int j = 0; j < 8; ++j) {
    int o = o0 + 8 * hi + j;
    yws[(size_t)n * 192 * SS + (size_t)o * SS + s0 + col] = acc[j] + bo[o];
  }
}

// ---------------------------------------------------------------------------
// Kernel 5: LSTM-style gates -> (h_next, m_next) concatenated in d_out (f32)
// ---------------------------------------------------------------------------
__global__ __launch_bounds__(256)
void gates_kernel(const float* __restrict__ yws, const float* __restrict__ mIn,
                  float* __restrict__ out)
{
  size_t idx = (size_t)blockIdx.x * 256 + threadIdx.x;
  const size_t TOT = (size_t)NB * 64 * SS;
  if (idx >= TOT) return;
  size_t n = idx / ((size_t)64 * SS);
  size_t rem = idx - n * 64 * SS;
  size_t yb = n * 192 * SS + rem;
  float iv = yws[yb];
  float gv = yws[yb + (size_t)64 * SS];
  float ov = yws[yb + (size_t)128 * SS];
  float mm = mIn[idx];
  float is = 1.f / (1.f + __expf(-iv));
  float gt = tanhf(gv);
  float mn = is * gt + (1.f - is) * mm;
  float hn = (1.f / (1.f + __expf(-ov))) * mn;
  out[idx] = hn;
  out[TOT + idx] = mn;
}

// ---------------------------------------------------------------------------
extern "C" void kernel_launch(void* const* d_in, const int* in_sizes, int n_in,
                              void* d_out, int out_size, void* d_ws, size_t ws_size,
                              hipStream_t stream)
{
  const float* h  = (const float*)d_in[0];
  const float* m  = (const float*)d_in[1];
  const float* Wh = (const float*)d_in[2];
  const float* bh = (const float*)d_in[3];
  const float* Wm = (const float*)d_in[4];
  const float* bm = (const float*)d_in[5];
  const float* Wz = (const float*)d_in[6];
  const float* bz = (const float*)d_in[7];
  const float* Wo = (const float*)d_in[8];
  const float* bo = (const float*)d_in[9];
  float* out = (float*)d_out;
  (void)in_sizes; (void)n_in; (void)out_size; (void)ws_size;

  char* ws = (char*)d_ws;
  size_t off = 0;
  auto take = [&](size_t bytes) -> char* {
    char* p = ws + off;
    off = (off + bytes + 255) & ~(size_t)255;
    return p;
  };
  h16*   qT  = (h16*)take((size_t)NB * SS * 64 * 2);
  h16*   khT = (h16*)take((size_t)NB * SS * 64 * 2);
  h16*   vhB = (h16*)take((size_t)NB * 64 * SS * 2);
  h16*   kmT = (h16*)take((size_t)NB * SS * 64 * 2);
  h16*   vmB = (h16*)take((size_t)NB * 64 * SS * 2);
  h16*   ZT  = (h16*)take((size_t)NB * SS * 128 * 2);
  h16*   xT  = (h16*)take((size_t)NB * SS * 128 * 2);
  float* yw  = (float*)take((size_t)NB * 192 * SS * 4);
  h16*   WzH = (h16*)take((size_t)64 * 128 * 2);
  h16*   WoR = (h16*)take((size_t)192 * 1152 * 2);

  hipLaunchKernelGGL(wprep_kernel, dim3(864), dim3(256), 0, stream,
                     Wz, Wo, WzH, WoR);
  hipLaunchKernelGGL(proj_kernel, dim3(128), dim3(256), 0, stream,
                     h, m, Wh, bh, Wm, bm, qT, khT, vhB, kmT, vmB, xT);
  hipLaunchKernelGGL(attn_kernel, dim3(1024), dim3(128), 0, stream,
                     qT, khT, vhB, kmT, vmB, ZT);
  hipLaunchKernelGGL(convz_kernel, dim3(2048), dim3(128), 0, stream,
                     WzH, bz, ZT, xT);
  hipLaunchKernelGGL(conv3_kernel, dim3(6144), dim3(128), 0, stream,
                     WoR, bo, xT, yw);
  hipLaunchKernelGGL(gates_kernel, dim3((NB * 64 * SS + 255) / 256), dim3(256), 0, stream,
                     yw, m, out);
}
